// transformerlayer_30219389895025
// MI455X (gfx1250) — compile-verified, matrix operand order checked by reading, not run on hardware
//
#include <hip/hip_runtime.h>
#include <hip/hip_bf16.h>

#define N_SEQ 512
#define DMODEL 128
#define LN_EPS 1e-5f

typedef _Float16 v16h __attribute__((ext_vector_type(16)));
typedef _Float16 h8   __attribute__((ext_vector_type(8)));
typedef float    v8f  __attribute__((ext_vector_type(8)));
typedef float    f4   __attribute__((ext_vector_type(4)));

// Load a 16-half WMMA A/B fragment: elements 0..7 = 8 contiguous halves at p,
// elements 8..15 = 8 contiguous halves at p+16 (matches 16-bit 16x32 layout).
__device__ __forceinline__ v16h ldfrag(const _Float16* p) {
    h8 lo = *reinterpret_cast<const h8*>(p);
    h8 hi = *reinterpret_cast<const h8*>(p + 16);
    v16h r;
#pragma unroll
    for (int e = 0; e < 8; ++e) { r[e] = lo[e]; r[e + 8] = hi[e]; }
    return r;
}

// Async global->LDS copy of 32 KB (one 128x128 f16 matrix), 256 threads.
// Uses CDNA5 GLOBAL_LOAD_ASYNC_TO_LDS_B128 (ASYNCcnt-tracked, no VGPR staging).
__device__ __forceinline__ void async_stage_weights(const _Float16* __restrict__ g,
                                                    _Float16* __restrict__ sW,
                                                    int tid) {
    unsigned int lds0 = (unsigned int)(uintptr_t)sW + (unsigned int)tid * 16u;
    const char* src = (const char*)g + (size_t)tid * 16u;
#pragma unroll
    for (int t = 0; t < 8; ++t) {
        asm volatile("global_load_async_to_lds_b128 %0, %1, off"
                     :: "v"(lds0 + (unsigned int)t * 4096u),
                        "v"(src + (size_t)t * 4096)
                     : "memory");
    }
}

__device__ __forceinline__ void wait_async_done() {
    asm volatile("s_wait_asynccnt 0x0" ::: "memory");
}

// ---------------------------------------------------------------------------
// K0: convert Woe/We1/We2 (f32 [k][n]) -> f16 transposed [n][k]
// ---------------------------------------------------------------------------
__global__ __launch_bounds__(256) void wtrans_kernel(
    const float* __restrict__ Woe, const float* __restrict__ We1,
    const float* __restrict__ We2,
    _Float16* __restrict__ WoeT, _Float16* __restrict__ We1T,
    _Float16* __restrict__ We2T)
{
    int t = blockIdx.x * 256 + threadIdx.x;      // 0 .. 3*16384-1
    int m = t >> 14;
    int r = t & 16383;
    int k = r >> 7, nn = r & 127;
    const float* src = (m == 0) ? Woe : ((m == 1) ? We1 : We2);
    _Float16*    dst = (m == 0) ? WoeT : ((m == 1) ? We1T : We2T);
    dst[nn * DMODEL + k] = (_Float16)src[k * DMODEL + nn];
}

// ---------------------------------------------------------------------------
// K1: q/k/v projections. One wave per row, GEMV (tiny: 1024 rows).
// ---------------------------------------------------------------------------
__global__ __launch_bounds__(256) void qkv_kernel(
    const float* __restrict__ nodes,
    const float* __restrict__ Wq, const float* __restrict__ bq,
    const float* __restrict__ Wk, const float* __restrict__ bk,
    const float* __restrict__ Wv, const float* __restrict__ bv,
    float* __restrict__ qf, float* __restrict__ kf, float* __restrict__ vf)
{
    __shared__ float sX[8][DMODEL];
    const int w = threadIdx.x >> 5, lane = threadIdx.x & 31;
    const int row = blockIdx.x * 8 + w;          // 0..1023
#pragma unroll
    for (int p = 0; p < 4; ++p)
        sX[w][lane + 32 * p] = nodes[row * DMODEL + lane + 32 * p];
    __syncthreads();

    auto gemv = [&](const float* __restrict__ W, const float* __restrict__ bias,
                    float* __restrict__ out) {
        float acc[4];
#pragma unroll
        for (int p = 0; p < 4; ++p) acc[p] = bias[lane + 32 * p];
        for (int k = 0; k < DMODEL; ++k) {
            float xk = sX[w][k];
            const float* wr = W + k * DMODEL + lane;
#pragma unroll
            for (int p = 0; p < 4; ++p) acc[p] = fmaf(xk, wr[32 * p], acc[p]);
        }
#pragma unroll
        for (int p = 0; p < 4; ++p) out[row * DMODEL + lane + 32 * p] = acc[p];
    };
    gemv(Wq, bq, qf);
    gemv(Wk, bk, kf);
    gemv(Wv, bv, vf);
}

// ---------------------------------------------------------------------------
// K2: fused edge path. Workgroup = 16 i's x 8 j's = 128 output rows.
// Wave w owns j = J0+w, rows i = I0..I0+15.  Three WMMA GEMMs + 2 LN fused.
// Edge tensor is streamed with NT hints (268 MB > 192 MB L2); weights staged
// to LDS with async-to-LDS copies overlapped with the score phase.
// ---------------------------------------------------------------------------
__global__ __launch_bounds__(256) void edge_kernel(
    const float* __restrict__ Ef,
    const float* __restrict__ qf, const float* __restrict__ kf,
    const _Float16* __restrict__ WoeT, const _Float16* __restrict__ We1T,
    const _Float16* __restrict__ We2T,
    const float* __restrict__ boe, const float* __restrict__ be1,
    const float* __restrict__ be2,
    const float* __restrict__ g1, const float* __restrict__ b1,
    const float* __restrict__ g2, const float* __restrict__ b2,
    float* __restrict__ logits, float* __restrict__ eout)
{
    __shared__ _Float16 sW[DMODEL * DMODEL];       // 32 KB: weight [n][k]
    __shared__ _Float16 sA[8 * 16 * DMODEL];       // 32 KB: per-wave activations

    const int tid   = threadIdx.x;
    const int w     = tid >> 5;
    const int lane  = tid & 31;
    const int lmod  = lane & 15;
    const int lhalf = lane >> 4;

    const int wg = blockIdx.x;
    const int jt = wg & 63;
    const int it = (wg >> 6) & 31;
    const int b  = wg >> 11;
    const int I0 = it * 16;
    const int j  = jt * 8 + w;

    _Float16* sAw = sA + w * 16 * DMODEL;

    // residual base: Ef[b, j, I0+8*lhalf + e, nb*16+lmod]
    const float* residBase =
        Ef + ((b * N_SEQ + j) * N_SEQ + I0 + 8 * lhalf) * DMODEL + lmod;
#pragma unroll
    for (int e = 0; e < 8; ++e)
        __builtin_prefetch(residBase + e * DMODEL, 0, 1);

    // ---- Phase 0: async-stage Woe^T; compute scores s -> sAw (f16), logits ----
    async_stage_weights(WoeT, sW, tid);
    {
        const int m  = lane >> 1;
        const int i  = I0 + m;
        const int c0 = (lane & 1) * 64;
        const float* qrow = qf + (b * N_SEQ + i) * DMODEL + c0;
        const float* krow = kf + (b * N_SEQ + j) * DMODEL + c0;
        const float* erow = Ef + ((b * N_SEQ + i) * N_SEQ + j) * DMODEL + c0;
        _Float16* srow = sAw + m * DMODEL + c0;
        float hs[4] = {0.f, 0.f, 0.f, 0.f};
#pragma unroll
        for (int u = 0; u < 64; u += 4) {
            f4 qv = *(const f4*)(qrow + u);
            f4 kv = *(const f4*)(krow + u);
            f4 ev = __builtin_nontemporal_load((const f4*)(erow + u));
            float s0 = qv[0] * kv[0] * ev[0];
            float s1 = qv[1] * kv[1] * ev[1];
            float s2 = qv[2] * kv[2] * ev[2];
            float s3 = qv[3] * kv[3] * ev[3];
            srow[u + 0] = (_Float16)s0;
            srow[u + 1] = (_Float16)s1;
            srow[u + 2] = (_Float16)s2;
            srow[u + 3] = (_Float16)s3;
            hs[u >> 4] += s0 + s1 + s2 + s3;
        }
        const int h0 = c0 >> 4;
#pragma unroll
        for (int hh = 0; hh < 4; ++hh)
            logits[(((b * 8 + h0 + hh) * N_SEQ) + i) * N_SEQ + j] =
                hs[hh] * 0.25f;  // 1/sqrt(16)
    }
    wait_async_done();
    __syncthreads();

    // ---- GEMM1: s @ Woe ----
    v8f acc[8];
#pragma unroll
    for (int nb = 0; nb < 8; ++nb)
#pragma unroll
        for (int e = 0; e < 8; ++e) acc[nb][e] = 0.f;

#pragma unroll
    for (int kb = 0; kb < 4; ++kb) {
        v16h a = ldfrag(sAw + lmod * DMODEL + kb * 32 + lhalf * 8);
#pragma unroll
        for (int nb = 0; nb < 8; ++nb) {
            v16h bf = ldfrag(sW + (nb * 16 + lmod) * DMODEL + kb * 32 + lhalf * 8);
            acc[nb] = __builtin_amdgcn_wmma_f32_16x16x32_f16(
                false, a, false, bf, (short)0, acc[nb], false, false);
        }
    }

    // ---- y = resid + acc + boe ; LayerNorm 1 (in-register, shfl over 16 lanes)
    float treg[8][8];
    float rsum[8], rsq[8];
#pragma unroll
    for (int e = 0; e < 8; ++e) { rsum[e] = 0.f; rsq[e] = 0.f; }
#pragma unroll
    for (int nb = 0; nb < 8; ++nb) {
        float bv = boe[nb * 16 + lmod];
#pragma unroll
        for (int e = 0; e < 8; ++e) {
            float rv = __builtin_nontemporal_load(residBase + e * DMODEL + nb * 16);
            float y = rv + acc[nb][e] + bv;
            treg[nb][e] = y;
            rsum[e] += y;
            rsq[e]  += y * y;
        }
    }
#pragma unroll
    for (int off = 1; off < 16; off <<= 1)
#pragma unroll
        for (int e = 0; e < 8; ++e) {
            rsum[e] += __shfl_xor(rsum[e], off, 32);
            rsq[e]  += __shfl_xor(rsq[e],  off, 32);
        }
#pragma unroll
    for (int e = 0; e < 8; ++e) {
        float mu  = rsum[e] * (1.f / 128.f);
        float var = rsq[e] * (1.f / 128.f) - mu * mu;
        float rs  = rsqrtf(var + LN_EPS);
#pragma unroll
        for (int nb = 0; nb < 8; ++nb) {
            int nn = nb * 16 + lmod;
            float t = (treg[nb][e] - mu) * rs * g1[nn] + b1[nn];
            treg[nb][e] = t;                       // keep for residual of FFN
            sAw[(e + 8 * lhalf) * DMODEL + nn] = (_Float16)t;
        }
    }
    __syncthreads();

    // ---- async-stage We1^T ----
    async_stage_weights(We1T, sW, tid);
    wait_async_done();
    __syncthreads();

    // ---- GEMM2: t @ We1, relu ----
#pragma unroll
    for (int nb = 0; nb < 8; ++nb)
#pragma unroll
        for (int e = 0; e < 8; ++e) acc[nb][e] = 0.f;
#pragma unroll
    for (int kb = 0; kb < 4; ++kb) {
        v16h a = ldfrag(sAw + lmod * DMODEL + kb * 32 + lhalf * 8);
#pragma unroll
        for (int nb = 0; nb < 8; ++nb) {
            v16h bf = ldfrag(sW + (nb * 16 + lmod) * DMODEL + kb * 32 + lhalf * 8);
            acc[nb] = __builtin_amdgcn_wmma_f32_16x16x32_f16(
                false, a, false, bf, (short)0, acc[nb], false, false);
        }
    }
#pragma unroll
    for (int nb = 0; nb < 8; ++nb) {
        float bv = be1[nb * 16 + lmod];
#pragma unroll
        for (int e = 0; e < 8; ++e) {
            float hv = fmaxf(acc[nb][e] + bv, 0.f);
            sAw[(e + 8 * lhalf) * DMODEL + nb * 16 + lmod] = (_Float16)hv;
        }
    }
    __syncthreads();

    // ---- async-stage We2^T ----
    async_stage_weights(We2T, sW, tid);
    wait_async_done();
    __syncthreads();

    // ---- GEMM3: relu(h) @ We2 ----
#pragma unroll
    for (int nb = 0; nb < 8; ++nb)
#pragma unroll
        for (int e = 0; e < 8; ++e) acc[nb][e] = 0.f;
#pragma unroll
    for (int kb = 0; kb < 4; ++kb) {
        v16h a = ldfrag(sAw + lmod * DMODEL + kb * 32 + lhalf * 8);
#pragma unroll
        for (int nb = 0; nb < 8; ++nb) {
            v16h bf = ldfrag(sW + (nb * 16 + lmod) * DMODEL + kb * 32 + lhalf * 8);
            acc[nb] = __builtin_amdgcn_wmma_f32_16x16x32_f16(
                false, a, false, bf, (short)0, acc[nb], false, false);
        }
    }

    // ---- u = t + acc + be2 ; LayerNorm 2 ; NT store ----
#pragma unroll
    for (int e = 0; e < 8; ++e) { rsum[e] = 0.f; rsq[e] = 0.f; }
#pragma unroll
    for (int nb = 0; nb < 8; ++nb) {
        float bv = be2[nb * 16 + lmod];
#pragma unroll
        for (int e = 0; e < 8; ++e) {
            float u = treg[nb][e] + acc[nb][e] + bv;
            treg[nb][e] = u;
            rsum[e] += u;
            rsq[e]  += u * u;
        }
    }
#pragma unroll
    for (int off = 1; off < 16; off <<= 1)
#pragma unroll
        for (int e = 0; e < 8; ++e) {
            rsum[e] += __shfl_xor(rsum[e], off, 32);
            rsq[e]  += __shfl_xor(rsq[e],  off, 32);
        }
    float* outBase =
        eout + ((b * N_SEQ + j) * N_SEQ + I0 + 8 * lhalf) * DMODEL + lmod;
#pragma unroll
    for (int e = 0; e < 8; ++e) {
        float mu  = rsum[e] * (1.f / 128.f);
        float var = rsq[e] * (1.f / 128.f) - mu * mu;
        float rs  = rsqrtf(var + LN_EPS);
#pragma unroll
        for (int nb = 0; nb < 8; ++nb) {
            int nn = nb * 16 + lmod;
            float o = (treg[nb][e] - mu) * rs * g2[nn] + b2[nn];
            __builtin_nontemporal_store(o, outBase + e * DMODEL + nb * 16);
        }
    }
}

// ---------------------------------------------------------------------------
// K3: softmax over j + ctx = attn @ v. One wave per (b,h,i) row.
// ---------------------------------------------------------------------------
__global__ __launch_bounds__(256) void softmax_ctx_kernel(
    const float* __restrict__ logits, const float* __restrict__ vf,
    float* __restrict__ ctx)
{
    __shared__ float sPart[8][32][16];
    const int w = threadIdx.x >> 5, lane = threadIdx.x & 31;
    const int row = blockIdx.x * 8 + w;            // 0 .. B*H*n-1
    const int i = row & 511;
    const int h = (row >> 9) & 7;
    const int b = row >> 12;
    const float* L = logits + (long)row * N_SEQ;

    float p[16];
    float mx = -3.4e38f;
#pragma unroll
    for (int t = 0; t < 16; ++t) {
        p[t] = L[lane + 32 * t];
        mx = fmaxf(mx, p[t]);
    }
#pragma unroll
    for (int off = 16; off >= 1; off >>= 1) mx = fmaxf(mx, __shfl_xor(mx, off, 32));
    float se = 0.f;
#pragma unroll
    for (int t = 0; t < 16; ++t) { p[t] = __expf(p[t] - mx); se += p[t]; }
#pragma unroll
    for (int off = 16; off >= 1; off >>= 1) se += __shfl_xor(se, off, 32);

    float c[16];
#pragma unroll
    for (int d = 0; d < 16; ++d) c[d] = 0.f;
#pragma unroll
    for (int t = 0; t < 16; ++t) {
        const int jj = lane + 32 * t;
        const float* vrow = vf + (b * N_SEQ + jj) * DMODEL + h * 16;
        f4 v0 = *(const f4*)(vrow + 0);
        f4 v1 = *(const f4*)(vrow + 4);
        f4 v2 = *(const f4*)(vrow + 8);
        f4 v3 = *(const f4*)(vrow + 12);
        float pw = p[t];
        c[0]  = fmaf(pw, v0[0], c[0]);  c[1]  = fmaf(pw, v0[1], c[1]);
        c[2]  = fmaf(pw, v0[2], c[2]);  c[3]  = fmaf(pw, v0[3], c[3]);
        c[4]  = fmaf(pw, v1[0], c[4]);  c[5]  = fmaf(pw, v1[1], c[5]);
        c[6]  = fmaf(pw, v1[2], c[6]);  c[7]  = fmaf(pw, v1[3], c[7]);
        c[8]  = fmaf(pw, v2[0], c[8]);  c[9]  = fmaf(pw, v2[1], c[9]);
        c[10] = fmaf(pw, v2[2], c[10]); c[11] = fmaf(pw, v2[3], c[11]);
        c[12] = fmaf(pw, v3[0], c[12]); c[13] = fmaf(pw, v3[1], c[13]);
        c[14] = fmaf(pw, v3[2], c[14]); c[15] = fmaf(pw, v3[3], c[15]);
    }
#pragma unroll
    for (int d = 0; d < 16; ++d) sPart[w][lane][d] = c[d];
    __syncthreads();
    const int d = lane & 15;
    float out = 0.f;
#pragma unroll
    for (int l2 = 0; l2 < 32; ++l2) out += sPart[w][l2][d];
    if (lane < 16)
        ctx[(b * N_SEQ + i) * DMODEL + h * 16 + d] = out / se;
}

// ---------------------------------------------------------------------------
// K4: node output path (tiny). One wave per row.
// ---------------------------------------------------------------------------
__global__ __launch_bounds__(256) void nodepost_kernel(
    const float* __restrict__ nodes, const float* __restrict__ ctx,
    const float* __restrict__ Woh, const float* __restrict__ boh,
    const float* __restrict__ Wh1, const float* __restrict__ bh1,
    const float* __restrict__ Wh2, const float* __restrict__ bh2,
    const float* __restrict__ g1, const float* __restrict__ b1,
    const float* __restrict__ g2, const float* __restrict__ b2,
    float* __restrict__ xout)
{
    __shared__ float sX[8][DMODEL];
    __shared__ float sH[8][2 * DMODEL];
    const int w = threadIdx.x >> 5, lane = threadIdx.x & 31;
    const int row = blockIdx.x * 8 + w;            // 0..1023

#pragma unroll
    for (int p = 0; p < 4; ++p)
        sX[w][lane + 32 * p] = ctx[row * DMODEL + lane + 32 * p];
    __syncthreads();

    float y[4];
#pragma unroll
    for (int p = 0; p < 4; ++p)
        y[p] = nodes[row * DMODEL + lane + 32 * p] + boh[lane + 32 * p];
    for (int k = 0; k < DMODEL; ++k) {
        float xk = sX[w][k];
        const float* wr = Woh + k * DMODEL + lane;
#pragma unroll
        for (int p = 0; p < 4; ++p) y[p] = fmaf(xk, wr[32 * p], y[p]);
    }
    // LN1
    float s = 0.f, q = 0.f;
#pragma unroll
    for (int p = 0; p < 4; ++p) { s += y[p]; q += y[p] * y[p]; }
#pragma unroll
    for (int off = 16; off >= 1; off >>= 1) {
        s += __shfl_xor(s, off, 32);
        q += __shfl_xor(q, off, 32);
    }
    {
        float mu = s * (1.f / 128.f);
        float var = q * (1.f / 128.f) - mu * mu;
        float rs = rsqrtf(var + LN_EPS);
#pragma unroll
        for (int p = 0; p < 4; ++p) {
            int nn = lane + 32 * p;
            y[p] = (y[p] - mu) * rs * g1[nn] + b1[nn];
        }
    }
    __syncthreads();
#pragma unroll
    for (int p = 0; p < 4; ++p) sX[w][lane + 32 * p] = y[p];
    __syncthreads();

    // FFN: 128 -> 256 (relu) -> 128
    float hacc[8];
#pragma unroll
    for (int p = 0; p < 8; ++p) hacc[p] = bh1[lane + 32 * p];
    for (int k = 0; k < DMODEL; ++k) {
        float xk = sX[w][k];
        const float* wr = Wh1 + k * 2 * DMODEL + lane;
#pragma unroll
        for (int p = 0; p < 8; ++p) hacc[p] = fmaf(xk, wr[32 * p], hacc[p]);
    }
#pragma unroll
    for (int p = 0; p < 8; ++p) sH[w][lane + 32 * p] = fmaxf(hacc[p], 0.f);
    __syncthreads();

    float z[4];
#pragma unroll
    for (int p = 0; p < 4; ++p) z[p] = y[p] + bh2[lane + 32 * p];
    for (int k = 0; k < 2 * DMODEL; ++k) {
        float hk = sH[w][k];
        const float* wr = Wh2 + k * DMODEL + lane;
#pragma unroll
        for (int p = 0; p < 4; ++p) z[p] = fmaf(hk, wr[32 * p], z[p]);
    }
    // LN2
    s = 0.f; q = 0.f;
#pragma unroll
    for (int p = 0; p < 4; ++p) { s += z[p]; q += z[p] * z[p]; }
#pragma unroll
    for (int off = 16; off >= 1; off >>= 1) {
        s += __shfl_xor(s, off, 32);
        q += __shfl_xor(q, off, 32);
    }
    {
        float mu = s * (1.f / 128.f);
        float var = q * (1.f / 128.f) - mu * mu;
        float rs = rsqrtf(var + LN_EPS);
#pragma unroll
        for (int p = 0; p < 4; ++p) {
            int nn = lane + 32 * p;
            xout[row * DMODEL + nn] = (z[p] - mu) * rs * g2[nn] + b2[nn];
        }
    }
}

// ---------------------------------------------------------------------------
extern "C" void kernel_launch(void* const* d_in, const int* in_sizes, int n_in,
                              void* d_out, int out_size, void* d_ws, size_t ws_size,
                              hipStream_t stream) {
    const float* nodes = (const float*)d_in[0];
    const float* edges = (const float*)d_in[1];
    const float* Wq  = (const float*)d_in[2];  const float* bq  = (const float*)d_in[3];
    const float* Wk  = (const float*)d_in[4];  const float* bk  = (const float*)d_in[5];
    const float* Wv  = (const float*)d_in[6];  const float* bv  = (const float*)d_in[7];
    const float* Woh = (const float*)d_in[8];  const float* boh = (const float*)d_in[9];
    const float* Woe = (const float*)d_in[10]; const float* boe = (const float*)d_in[11];
    const float* Wh1 = (const float*)d_in[12]; const float* bh1 = (const float*)d_in[13];
    const float* Wh2 = (const float*)d_in[14]; const float* bh2 = (const float*)d_in[15];
    const float* We1 = (const float*)d_in[16]; const float* be1 = (const float*)d_in[17];
    const float* We2 = (const float*)d_in[18]; const float* be2 = (const float*)d_in[19];
    const float* g_hn1 = (const float*)d_in[20]; const float* b_hn1 = (const float*)d_in[21];
    const float* g_hn2 = (const float*)d_in[22]; const float* b_hn2 = (const float*)d_in[23];
    const float* g_en1 = (const float*)d_in[24]; const float* b_en1 = (const float*)d_in[25];
    const float* g_en2 = (const float*)d_in[26]; const float* b_en2 = (const float*)d_in[27];

    float* xout = (float*)d_out;
    float* eout = xout + 2 * N_SEQ * DMODEL;       // 131072 floats for x

    // workspace layout
    float* qf     = (float*)d_ws;                  // 131072
    float* kf     = qf + 131072;
    float* vf     = kf + 131072;
    float* ctx    = vf + 131072;
    float* logits = ctx + 131072;                  // 2*8*512*512 = 4194304
    _Float16* WoeT = (_Float16*)(logits + 4194304);
    _Float16* We1T = WoeT + DMODEL * DMODEL;
    _Float16* We2T = We1T + DMODEL * DMODEL;

    wtrans_kernel<<<192, 256, 0, stream>>>(Woe, We1, We2, WoeT, We1T, We2T);
    qkv_kernel<<<128, 256, 0, stream>>>(nodes, Wq, bq, Wk, bk, Wv, bv, qf, kf, vf);
    edge_kernel<<<4096, 256, 0, stream>>>(edges, qf, kf, WoeT, We1T, We2T,
                                          boe, be1, be2, g_en1, b_en1, g_en2, b_en2,
                                          logits, eout);
    softmax_ctx_kernel<<<1024, 256, 0, stream>>>(logits, vf, ctx);
    nodepost_kernel<<<128, 256, 0, stream>>>(nodes, ctx, Woh, boh, Wh1, bh1, Wh2, bh2,
                                             g_hn1, b_hn1, g_hn2, b_hn2, xout);
}
